// HeteroDCRNN_4449586119221
// MI455X (gfx1250) — compile-verified, matrix operand order checked by reading, not run on hardware
//
#include <hip/hip_runtime.h>
#include <hip/hip_bf16.h>
#include <math.h>

typedef __attribute__((ext_vector_type(16))) _Float16 v16h;
typedef __attribute__((ext_vector_type(8)))  _Float16 v8h;
typedef __attribute__((ext_vector_type(8)))  float    v8f;

namespace {
constexpr int B_   = 128;
constexpr int T_   = 24;
constexpr int NS   = 20;
constexpr int NP   = 8;
constexpr int NALL = 28;          // NS + NP
constexpr int F_   = 256;
constexpr int FO   = 256;
constexpr int FIN  = 512;         // F + FO
constexpr int KS   = 5 * FIN;     // stream stacked K (X, T1o, T1i, T2o, T2i)
constexpr int KP   = 3 * FIN;     // precip stacked K (X, T1o, T1i)
constexpr int MS   = B_ * NS;     // 2560
constexpr int MP   = B_ * NP;     // 1024
constexpr size_t T1STRIDE = (size_t)MS * FIN;  // per-direction T1 scratch stride
}

// ---------------------------------------------------------------------------
// One-time prep: CSR-by-destination edge lists + per-edge diffusion coefs.
// c_o[e] = 1/deg_out[row[e]], c_i[e] = 1/deg_in[row[e]]  (matches reference:
// norm arrays are both indexed at `row`).  E<=100, single thread is fine.
// ---------------------------------------------------------------------------
__global__ void build_csr_k(const int* __restrict__ ei, const float* __restrict__ ew,
                            int E, int N,
                            int* __restrict__ ptr, int* __restrict__ srcn,
                            float* __restrict__ cco, float* __restrict__ cci) {
  if (threadIdx.x != 0 || blockIdx.x != 0) return;
  float dout[NS + 1], din[NS + 1];
  for (int n = 0; n < N; ++n) { dout[n] = 0.f; din[n] = 0.f; }
  const int* row = ei;
  const int* col = ei + E;
  for (int e = 0; e < E; ++e) { dout[row[e]] += ew[e]; din[col[e]] += ew[e]; }
  for (int n = 0; n <= N; ++n) ptr[n] = 0;
  for (int e = 0; e < E; ++e) ptr[col[e] + 1] += 1;
  for (int n = 0; n < N; ++n) ptr[n + 1] += ptr[n];
  int fill[NS + 1];
  for (int n = 0; n < N; ++n) fill[n] = ptr[n];
  for (int e = 0; e < E; ++e) {
    int c = col[e];
    int p = fill[c]++;
    srcn[p] = row[e];
    cco[p] = 1.f / dout[row[e]];
    cci[p] = 1.f / din[row[e]];
  }
}

// ---------------------------------------------------------------------------
// One-time prep: stack gate weights along K and transpose to N-major f16.
// Term order: 0 -> W[0,0]+W[1,0] ; 1 -> W[0,1] ; 2 -> W[1,1] ; 3 -> W[0,2] ;
// 4 -> W[1,2].  Source layout: W[2][Kcheb][FIN][FO].
// ---------------------------------------------------------------------------
__global__ void prep_w_k(const float* __restrict__ Wz, const float* __restrict__ Wr,
                         const float* __restrict__ Wh, _Float16* __restrict__ out,
                         int Kcheb, int Ktot) {
  size_t idx = (size_t)blockIdx.x * blockDim.x + threadIdx.x;
  size_t total = (size_t)3 * FO * Ktot;
  if (idx >= total) return;
  int kk = (int)(idx % Ktot);
  int n  = (int)((idx / Ktot) % FO);
  int g  = (int)(idx / ((size_t)Ktot * FO));
  const float* W = (g == 0) ? Wz : (g == 1) ? Wr : Wh;
  int term = kk / FIN, fin = kk % FIN;
  float v;
  if (term == 0) {
    v = W[(((size_t)0 * Kcheb + 0) * FIN + fin) * FO + n] +
        W[(((size_t)1 * Kcheb + 0) * FIN + fin) * FO + n];
  } else {
    int k = (term + 1) / 2;
    int d = (term + 1) % 2;
    v = W[(((size_t)d * Kcheb + k) * FIN + fin) * FO + n];
  }
  out[(size_t)g * FO * Ktot + (size_t)n * Ktot + kk] = (_Float16)v;
}

// ---------------------------------------------------------------------------
// Per-step pack: builds f16 A-matrix columns for terms 0 (identity), 1 (T1_o),
// 2 (T1_i) of concat(X_t, Hsrc).  Feature columns f<256 come from X, f>=256
// from Hsrc; H-phase passes fbase=256 so only H columns are rewritten.
// ---------------------------------------------------------------------------
__global__ void pack01_k(const float* __restrict__ Xall, int t, int NN,
                         const float* __restrict__ Hsrc, int nodeOff,
                         const int* __restrict__ ptr, const int* __restrict__ srcn,
                         const float* __restrict__ cco, const float* __restrict__ cci,
                         _Float16* __restrict__ A, int Ktot, float* __restrict__ T1,
                         int fbase, int fcount) {
  int idx = blockIdx.x * blockDim.x + threadIdx.x;
  int total = B_ * NN * fcount;
  if (idx >= total) return;
  int f = fbase + idx % fcount;
  int n = (idx / fcount) % NN;
  int b = idx / (fcount * NN);
  auto val = [&](int m) -> float {
    if (f < F_) return Xall[(((size_t)b * T_ + t) * NN + m) * F_ + f];
    return Hsrc[((size_t)b * NALL + nodeOff + m) * FO + (f - F_)];
  };
  float v0 = val(n);
  float t1o = 0.f, t1i = 0.f;
  int j1 = ptr[n + 1];
  for (int j = ptr[n]; j < j1; ++j) {
    float vm = val(srcn[j]);
    t1o += cco[j] * vm;
    t1i += cci[j] * vm;
  }
  _Float16* Ar = A + (size_t)(b * NN + n) * Ktot;
  Ar[f]           = (_Float16)v0;
  Ar[FIN + f]     = (_Float16)t1o;
  Ar[2 * FIN + f] = (_Float16)t1i;
  if (T1) {
    size_t o = (size_t)(b * NN + n) * FIN + f;
    T1[o] = t1o;
    T1[T1STRIDE + o] = t1i;
  }
}

// Chebyshev order-2 terms (stream graph only): T2 = 2*prop(T1) - T0.
__global__ void pack2_k(const float* __restrict__ Xall, int t,
                        const float* __restrict__ Hsrc,
                        const int* __restrict__ ptr, const int* __restrict__ srcn,
                        const float* __restrict__ cco, const float* __restrict__ cci,
                        _Float16* __restrict__ A, const float* __restrict__ T1,
                        int fbase, int fcount) {
  int idx = blockIdx.x * blockDim.x + threadIdx.x;
  int total = B_ * NS * fcount;
  if (idx >= total) return;
  int f = fbase + idx % fcount;
  int n = (idx / fcount) % NS;
  int b = idx / (fcount * NS);
  float v0 = (f < F_) ? Xall[(((size_t)b * T_ + t) * NS + n) * F_ + f]
                      : Hsrc[((size_t)b * NALL + n) * FO + (f - F_)];
  float a2o = 0.f, a2i = 0.f;
  int j1 = ptr[n + 1];
  for (int j = ptr[n]; j < j1; ++j) {
    size_t o = (size_t)(b * NS + srcn[j]) * FIN + f;
    a2o += cco[j] * T1[o];
    a2i += cci[j] * T1[T1STRIDE + o];
  }
  _Float16* Ar = A + (size_t)(b * NS + n) * KS;
  Ar[3 * FIN + f] = (_Float16)(2.f * a2o - v0);
  Ar[4 * FIN + f] = (_Float16)(2.f * a2i - v0);
}

// ---------------------------------------------------------------------------
// Fragment load helper: ISA 16-bit A/B layout for 16x16x32 f16 WMMA.
// Lanes 0-15 hold K k0..k0+7 (lo) and k0+16..k0+23 (hi); lanes 16-31 +8.
// khalf is folded into the base pointer by the caller.
// ---------------------------------------------------------------------------
__device__ __forceinline__ v16h ldfrag(const _Float16* p, int k0) {
  v8h lo = *(const v8h*)(p + k0);
  v8h hi = *(const v8h*)(p + k0 + 16);
  return __builtin_shufflevector(lo, hi, 0, 1, 2, 3, 4, 5, 6, 7,
                                 8, 9, 10, 11, 12, 13, 14, 15);
}

__device__ __forceinline__ v8f wmma_f16(v16h a, v16h b, v8f c) {
  return __builtin_amdgcn_wmma_f32_16x16x32_f16(
      /*neg_a=*/false, a, /*neg_b=*/false, b,
      /*c_mod=*/(short)0, c, /*reuse_a=*/false, /*reuse_b=*/false);
}

// ---------------------------------------------------------------------------
// WMMA GEMM: C[b, nodeOff+n, :] = act(A(MxK) * Wt(256xK, N-major)^T + bias).
// One wave computes a 32x64 strip (2 A frags x 4 B frags -> 8 WMMAs per
// 32-deep K chunk), with register double-buffering so the next chunk's loads
// overlap the current chunk's WMMAs (waits become loadcnt<=N, not 0).
// Grid is sized exactly (M % 32 == 0), so EXEC is all-ones for every WMMA.
// ACT: 0=none, 1=sigmoid, 2=tanh.
// ---------------------------------------------------------------------------
template <int ACT>
__global__ void gemm_wmma_k(const _Float16* __restrict__ A,
                            const _Float16* __restrict__ Wt,
                            const float* __restrict__ bias,
                            float* __restrict__ C,
                            int M, int Ktot, int nodesPer, int nodeOff) {
  int gwave = (int)((blockIdx.x * (size_t)blockDim.x + threadIdx.x) >> 5);
  int lane = threadIdx.x & 31;
  int tm = gwave >> 2;                 // 32-row tile
  int tn4 = gwave & 3;                 // 64-col tile
  int rlane = lane & 15;
  int khalf = (lane >> 4) * 8;         // ISA 16-bit layout: lanes16-31 hold K+8

  const _Float16* A0 = A + (size_t)(tm * 32 + rlane) * Ktot + khalf;
  const _Float16* A1 = A0 + (size_t)16 * Ktot;
  const _Float16* W0 = Wt + (size_t)(tn4 * 64 + rlane) * Ktot + khalf;
  const _Float16* W1 = W0 + (size_t)16 * Ktot;
  const _Float16* W2 = W0 + (size_t)32 * Ktot;
  const _Float16* W3 = W0 + (size_t)48 * Ktot;

  v8f acc[8] = {};

  // software pipeline: prime chunk 0
  v16h a0 = ldfrag(A0, 0), a1 = ldfrag(A1, 0);
  v16h b0 = ldfrag(W0, 0), b1 = ldfrag(W1, 0);
  v16h b2 = ldfrag(W2, 0), b3 = ldfrag(W3, 0);

  for (int k0 = 32; k0 < Ktot; k0 += 32) {
    // issue next chunk's loads before consuming the current fragments
    v16h na0 = ldfrag(A0, k0), na1 = ldfrag(A1, k0);
    v16h nb0 = ldfrag(W0, k0), nb1 = ldfrag(W1, k0);
    v16h nb2 = ldfrag(W2, k0), nb3 = ldfrag(W3, k0);

    acc[0] = wmma_f16(a0, b0, acc[0]);
    acc[1] = wmma_f16(a0, b1, acc[1]);
    acc[2] = wmma_f16(a0, b2, acc[2]);
    acc[3] = wmma_f16(a0, b3, acc[3]);
    acc[4] = wmma_f16(a1, b0, acc[4]);
    acc[5] = wmma_f16(a1, b1, acc[5]);
    acc[6] = wmma_f16(a1, b2, acc[6]);
    acc[7] = wmma_f16(a1, b3, acc[7]);

    a0 = na0; a1 = na1;
    b0 = nb0; b1 = nb1; b2 = nb2; b3 = nb3;
  }
  // epilogue chunk
  acc[0] = wmma_f16(a0, b0, acc[0]);
  acc[1] = wmma_f16(a0, b1, acc[1]);
  acc[2] = wmma_f16(a0, b2, acc[2]);
  acc[3] = wmma_f16(a0, b3, acc[3]);
  acc[4] = wmma_f16(a1, b0, acc[4]);
  acc[5] = wmma_f16(a1, b1, acc[5]);
  acc[6] = wmma_f16(a1, b2, acc[6]);
  acc[7] = wmma_f16(a1, b3, acc[7]);

  // C layout: VGPR v -> row v (lanes 0-15) / row v+8 (lanes 16-31); col = lane&15
#pragma unroll
  for (int i = 0; i < 2; ++i) {
    int rbase = tm * 32 + i * 16 + ((lane >> 4) << 3);
#pragma unroll
    for (int s = 0; s < 4; ++s) {
      int col = tn4 * 64 + s * 16 + rlane;
      float bn = bias[col];
      v8f av = acc[i * 4 + s];
#pragma unroll
      for (int v = 0; v < 8; ++v) {
        int r = rbase + v;
        float x = av[v] + bn;
        if (ACT == 1) x = 1.f / (1.f + expf(-x));
        else if (ACT == 2) x = tanhf(x);
        int b = r / nodesPer, n = r - b * nodesPer;
        C[((size_t)b * NALL + nodeOff + n) * FO + col] = x;
      }
    }
  }
}

// ---------------------------------------------------------------------------
// GRU elementwise pieces.
// ---------------------------------------------------------------------------
__global__ void zero_k(float* __restrict__ p, int n) {
  int i = blockIdx.x * blockDim.x + threadIdx.x;
  if (i < n) p[i] = 0.f;
}

__global__ void gh_k(const float* __restrict__ R, const float* __restrict__ H,
                     float* __restrict__ gH, int n) {
  int i = blockIdx.x * blockDim.x + threadIdx.x;
  if (i < n) gH[i] = R[i] * H[i];
}

__global__ void update_k(const float* __restrict__ Z, const float* __restrict__ Ht,
                         float* __restrict__ H, int n) {
  int i = blockIdx.x * blockDim.x + threadIdx.x;
  if (i < n) {
    float z = Z[i];
    H[i] = z * H[i] + (1.f - z) * Ht[i];
  }
}

// ---------------------------------------------------------------------------
// Readout: ro[b,n] = H[b,n,:]·W_ro + b_ro ; out[b,j] = ro[b,:]·W_ag[:,j]+b_ag.
// ---------------------------------------------------------------------------
__global__ void readout_k(const float* __restrict__ H, const float* __restrict__ Wro,
                          const float* __restrict__ bro, const float* __restrict__ Wag,
                          const float* __restrict__ bag, float* __restrict__ out) {
  __shared__ float red[256];
  __shared__ float ro[NALL];
  int b = blockIdx.x;
  int tid = threadIdx.x;
  for (int n = 0; n < NALL; ++n) {
    red[tid] = H[((size_t)b * NALL + n) * FO + tid] * Wro[tid];
    __syncthreads();
    for (int st = 128; st > 0; st >>= 1) {
      if (tid < st) red[tid] += red[tid + st];
      __syncthreads();
    }
    if (tid == 0) ro[n] = red[0] + bro[0];
    __syncthreads();
  }
  if (tid < 5) {
    float s = bag[tid];
    for (int n = 0; n < NALL; ++n) s += ro[n] * Wag[n * 5 + tid];
    out[b * 5 + tid] = s;
  }
}

// ---------------------------------------------------------------------------
extern "C" void kernel_launch(void* const* d_in, const int* in_sizes, int n_in,
                              void* d_out, int out_size, void* d_ws, size_t ws_size,
                              hipStream_t stream) {
  (void)n_in; (void)out_size; (void)ws_size;
  const float* x_dis = (const float*)d_in[0];
  const float* x_pre = (const float*)d_in[1];
  const int*   ei_s  = (const int*)d_in[2];
  const int*   ei_p  = (const int*)d_in[3];
  const float* ew_s  = (const float*)d_in[4];
  const float* ew_p  = (const float*)d_in[5];
  const float* Wz_s = (const float*)d_in[6];  const float* bz_s = (const float*)d_in[7];
  const float* Wr_s = (const float*)d_in[8];  const float* br_s = (const float*)d_in[9];
  const float* Wh_s = (const float*)d_in[10]; const float* bh_s = (const float*)d_in[11];
  const float* Wz_p = (const float*)d_in[12]; const float* bz_p = (const float*)d_in[13];
  const float* Wr_p = (const float*)d_in[14]; const float* br_p = (const float*)d_in[15];
  const float* Wh_p = (const float*)d_in[16]; const float* bh_p = (const float*)d_in[17];
  const float* W_ro = (const float*)d_in[18]; const float* b_ro = (const float*)d_in[19];
  const float* W_ag = (const float*)d_in[20]; const float* b_ag = (const float*)d_in[21];
  float* out = (float*)d_out;

  const int ES = in_sizes[2] / 2;   // 100
  const int EP = in_sizes[3] / 2;   // 32

  // Workspace carve-up (all 256B aligned; total ~52 MB).
  char* w = (char*)d_ws;
  auto carve = [&](size_t bytes) {
    char* p = w;
    w += (bytes + 255) & ~(size_t)255;
    return p;
  };
  _Float16* WstkS = (_Float16*)carve((size_t)3 * FO * KS * sizeof(_Float16));
  _Float16* WstkP = (_Float16*)carve((size_t)3 * FO * KP * sizeof(_Float16));
  _Float16* A_s   = (_Float16*)carve((size_t)MS * KS * sizeof(_Float16));
  _Float16* A_p   = (_Float16*)carve((size_t)MP * KP * sizeof(_Float16));
  float* T1 = (float*)carve(2 * T1STRIDE * sizeof(float));
  const int NHF = B_ * NALL * FO;
  float* H   = (float*)carve((size_t)NHF * sizeof(float));
  float* gH  = (float*)carve((size_t)NHF * sizeof(float));
  float* Z   = (float*)carve((size_t)NHF * sizeof(float));
  float* R   = (float*)carve((size_t)NHF * sizeof(float));
  float* Ht  = (float*)carve((size_t)NHF * sizeof(float));
  int*   ptrS = (int*)carve((NS + 1) * sizeof(int));
  int*   srcS = (int*)carve(ES * sizeof(int));
  int*   ptrP = (int*)carve((NP + 1) * sizeof(int));
  int*   srcP = (int*)carve(EP * sizeof(int));
  float* ccoS = (float*)carve(ES * sizeof(float));
  float* cciS = (float*)carve(ES * sizeof(float));
  float* ccoP = (float*)carve(EP * sizeof(float));
  float* cciP = (float*)carve(EP * sizeof(float));

  // ---- one-time prep ----
  build_csr_k<<<1, 1, 0, stream>>>(ei_s, ew_s, ES, NS, ptrS, srcS, ccoS, cciS);
  build_csr_k<<<1, 1, 0, stream>>>(ei_p, ew_p, EP, NP, ptrP, srcP, ccoP, cciP);
  prep_w_k<<<(3 * FO * KS + 255) / 256, 256, 0, stream>>>(Wz_s, Wr_s, Wh_s, WstkS, 3, KS);
  prep_w_k<<<(3 * FO * KP + 255) / 256, 256, 0, stream>>>(Wz_p, Wr_p, Wh_p, WstkP, 2, KP);
  zero_k<<<(NHF + 255) / 256, 256, 0, stream>>>(H, NHF);

  const int gPackSFull = (B_ * NS * FIN + 255) / 256;
  const int gPackSHalf = (B_ * NS * FO + 255) / 256;
  const int gPackPFull = (B_ * NP * FIN + 255) / 256;
  const int gPackPHalf = (B_ * NP * FO + 255) / 256;
  const int gElem = (NHF + 255) / 256;
  const int gGemmS = MS / 64;  // exact: (M/32 tiles)*(4 n-tiles)*32 lanes / 256
  const int gGemmP = MP / 64;

  // ---- recurrence ----
  for (int t = 0; t < T_; ++t) {
    // Z/R phase: A from concat(X_t, H)
    pack01_k<<<gPackSFull, 256, 0, stream>>>(x_dis, t, NS, H, 0, ptrS, srcS, ccoS, cciS,
                                             A_s, KS, T1, 0, FIN);
    pack2_k<<<gPackSFull, 256, 0, stream>>>(x_dis, t, H, ptrS, srcS, ccoS, cciS,
                                            A_s, T1, 0, FIN);
    pack01_k<<<gPackPFull, 256, 0, stream>>>(x_pre, t, NP, H, NS, ptrP, srcP, ccoP, cciP,
                                             A_p, KP, nullptr, 0, FIN);

    gemm_wmma_k<1><<<gGemmS, 256, 0, stream>>>(A_s, WstkS + (size_t)0 * FO * KS, bz_s, Z,
                                               MS, KS, NS, 0);
    gemm_wmma_k<1><<<gGemmP, 256, 0, stream>>>(A_p, WstkP + (size_t)0 * FO * KP, bz_p, Z,
                                               MP, KP, NP, NS);
    gemm_wmma_k<1><<<gGemmS, 256, 0, stream>>>(A_s, WstkS + (size_t)1 * FO * KS, br_s, R,
                                               MS, KS, NS, 0);
    gemm_wmma_k<1><<<gGemmP, 256, 0, stream>>>(A_p, WstkP + (size_t)1 * FO * KP, br_p, R,
                                               MP, KP, NP, NS);

    gh_k<<<gElem, 256, 0, stream>>>(R, H, gH, NHF);

    // Ht phase: only H-columns (f>=256) of A change (X terms are reused).
    pack01_k<<<gPackSHalf, 256, 0, stream>>>(x_dis, t, NS, gH, 0, ptrS, srcS, ccoS, cciS,
                                             A_s, KS, T1, F_, FO);
    pack2_k<<<gPackSHalf, 256, 0, stream>>>(x_dis, t, gH, ptrS, srcS, ccoS, cciS,
                                            A_s, T1, F_, FO);
    pack01_k<<<gPackPHalf, 256, 0, stream>>>(x_pre, t, NP, gH, NS, ptrP, srcP, ccoP, cciP,
                                             A_p, KP, nullptr, F_, FO);

    gemm_wmma_k<2><<<gGemmS, 256, 0, stream>>>(A_s, WstkS + (size_t)2 * FO * KS, bh_s, Ht,
                                               MS, KS, NS, 0);
    gemm_wmma_k<2><<<gGemmP, 256, 0, stream>>>(A_p, WstkP + (size_t)2 * FO * KP, bh_p, Ht,
                                               MP, KP, NP, NS);

    update_k<<<gElem, 256, 0, stream>>>(Z, Ht, H, NHF);
  }

  // ---- readout ----
  readout_k<<<B_, 256, 0, stream>>>(H, W_ro, b_ro, W_ag, b_ag, out);
}
